// SplineNet_3736621547797
// MI455X (gfx1250) — matrix operand order repaired and checked by reading, not compiled, not run on hardware
//
#include <hip/hip_runtime.h>
#include <cstdint>
#include <cmath>

// ---------------------------------------------------------------------------
// SplineNet on gfx1250 (MI455X, wave32).
//
// Dense GEMM parts (x@W[k], x@root, MLP) -> v_wmma_f32_16x16x32_f16
//   (f16 A/B fragments, f32 accumulate; D=32 => exactly two 16x16 N-tiles).
// Sparse part (edge gather + mean scatter) -> 1 wave per edge, 1 lane per
//   feature: 128B coalesced gathers of precomputed y rows, f32 atomics.
// Roofline: dominated by ~2GB of edge gather/scatter traffic; y1 (115MB)
//   is L2-resident (192MB), GEMM FLOPs (~15G) are negligible on WMMA.
// ---------------------------------------------------------------------------

typedef __attribute__((ext_vector_type(16))) _Float16 v16h;
typedef __attribute__((ext_vector_type(8)))  float    v8f;

#define DFEAT 32

__device__ __forceinline__ int imin(int a, int b) { return a < b ? a : b; }

// A-fragment K index for element (vgpr j, half h), per ISA 16-bit A 16x32 layout.
__device__ __forceinline__ int a_kidx(int j, int h, int laneHi) {
    int base = (j < 4) ? (2 * j + h) : (16 + 2 * (j - 4) + h);
    return base + laneHi * 8;
}
// B-fragment K index for element (vgpr j, half h), 16-bit B 32x16 layout.
__device__ __forceinline__ int b_kidx(int j, int h, int laneHi) {
    int base = (j < 4) ? (2 * j + h) : (16 + 2 * (j - 4) + h);
    return base + laneHi * 8;
}

// Load a 16x32 A fragment: rows node0..node0+15 of a row-major N x 32 f32 matrix.
__device__ __forceinline__ v16h load_a_f32(const float* __restrict__ src, int node0, int lane) {
    int m = lane & 15, hi = lane >> 4;
    v16h a;
#pragma unroll
    for (int j = 0; j < 8; ++j)
#pragma unroll
        for (int h = 0; h < 2; ++h)
            a[2 * j + h] = (_Float16)src[(size_t)(node0 + m) * DFEAT + a_kidx(j, h, hi)];
    return a;
}

// Load a 32x16 B fragment: columns nbase..nbase+15 of a row-major 32 x ncols f32 matrix.
__device__ __forceinline__ v16h load_b_f32(const float* __restrict__ w, int ncols, int nbase, int lane) {
    int m = lane & 15, hi = lane >> 4;
    v16h b;
#pragma unroll
    for (int j = 0; j < 8; ++j)
#pragma unroll
        for (int h = 0; h < 2; ++h)
            b[2 * j + h] = (_Float16)w[(size_t)b_kidx(j, h, hi) * ncols + nbase + m];
    return b;
}

__device__ __forceinline__ v8f wmma16(v16h a, v16h b, v8f c) {
    // (neg_a, A, neg_b, B, c_mod, C, reuse_a, reuse_b)
    return __builtin_amdgcn_wmma_f32_16x16x32_f16(false, a, false, b, (short)0, c, false, false);
}

// ---------------------------------------------------------------------------
// Kernel 1: y[n,k,:] = x[n] @ W[k]   (one wave per 16-node tile, loop over k)
// ---------------------------------------------------------------------------
__global__ void __launch_bounds__(64)
spline_y_kernel(const float* __restrict__ x, const float* __restrict__ W,
                float* __restrict__ y, int nTiles, int KK) {
    int wave = (blockIdx.x * blockDim.x + threadIdx.x) >> 5;
    if (wave >= nTiles) return;                 // wave-uniform: EXEC stays all-1s
    int lane = threadIdx.x & 31;
    int m = lane & 15, hi = lane >> 4;
    int node0 = wave * 16;

    v16h a = load_a_f32(x, node0, lane);        // reuse x tile across all k
    for (int k = 0; k < KK; ++k) {
        const float* wk = W + (size_t)k * DFEAT * DFEAT;
        v16h blo = load_b_f32(wk, DFEAT, 0, lane);
        v16h bhi = load_b_f32(wk, DFEAT, 16, lane);
        v8f c0 = {0.f, 0.f, 0.f, 0.f, 0.f, 0.f, 0.f, 0.f};
        v8f c1 = {0.f, 0.f, 0.f, 0.f, 0.f, 0.f, 0.f, 0.f};
        c0 = wmma16(a, blo, c0);
        c1 = wmma16(a, bhi, c1);
#pragma unroll
        for (int j = 0; j < 8; ++j) {           // D layout: row = j + 8*hi, col = m
            int node = node0 + j + hi * 8;
            float* out = y + ((size_t)node * KK + k) * DFEAT;
            out[m]      = c0[j];
            out[m + 16] = c1[j];
        }
    }
}

// ---------------------------------------------------------------------------
// Kernel 2: edge scatter.  One wave per edge, lane = feature.
//   msg[e] = sum over 4 bilinear taps  b_t * y[src, kidx_t, :]
//   agg[dst] += msg ; cnt[dst] += 1
// ---------------------------------------------------------------------------
__global__ void __launch_bounds__(128)
edge_scatter_kernel(const int* __restrict__ srcA, const int* __restrict__ dstA,
                    const float* __restrict__ attr, const float* __restrict__ y,
                    float* __restrict__ agg, float* __restrict__ cnt,
                    int E, int KK, int ksize) {
    int e = (blockIdx.x * blockDim.x + threadIdx.x) >> 5;
    if (e >= E) return;
    int lane = threadIdx.x & 31;

    int s = srcA[e];
    int d = dstA[e];
    float u = attr[(size_t)e * 2 + 0] * (float)(ksize - 1);
    float v = attr[(size_t)e * 2 + 1] * (float)(ksize - 1);
    float bu = floorf(u), bv = floorf(v);
    int   iu = (int)bu,   iv = (int)bv;
    float fu = u - bu,    fv = v - bv;

    float acc = 0.f;
#pragma unroll
    for (int tap = 0; tap < 4; ++tap) {
        int ou = tap & 1, ov = tap >> 1;
        float w = (ou ? fu : 1.f - fu) * (ov ? fv : 1.f - fv);
        int ku = imin(iu + ou, ksize - 1);
        int kv = imin(iv + ov, ksize - 1);
        int kidx = ku + kv * ksize;             // dim0 stride 1, dim1 stride ksize
        acc += w * y[((size_t)s * KK + kidx) * DFEAT + lane];   // 128B coalesced
    }
    atomicAdd(&agg[(size_t)d * DFEAT + lane], acc);
    if (lane == 0) atomicAdd(&cnt[d], 1.0f);
}

// ---------------------------------------------------------------------------
// Kernel 3: node update:  h[n] = ELU( agg[n]/max(cnt,1) + x[n]@root + bias )
// ---------------------------------------------------------------------------
__global__ void __launch_bounds__(64)
node_update_kernel(const float* __restrict__ xin, const float* __restrict__ root,
                   const float* __restrict__ bias, const float* __restrict__ agg,
                   const float* __restrict__ cnt, float* __restrict__ hout, int nTiles) {
    int wave = (blockIdx.x * blockDim.x + threadIdx.x) >> 5;
    if (wave >= nTiles) return;
    int lane = threadIdx.x & 31;
    int m = lane & 15, hi = lane >> 4;
    int node0 = wave * 16;

    v16h a   = load_a_f32(xin, node0, lane);
    v16h blo = load_b_f32(root, DFEAT, 0, lane);
    v16h bhi = load_b_f32(root, DFEAT, 16, lane);
    v8f c0 = {0.f, 0.f, 0.f, 0.f, 0.f, 0.f, 0.f, 0.f};
    v8f c1 = {0.f, 0.f, 0.f, 0.f, 0.f, 0.f, 0.f, 0.f};
    c0 = wmma16(a, blo, c0);
    c1 = wmma16(a, bhi, c1);

#pragma unroll
    for (int j = 0; j < 8; ++j) {
        int node = node0 + j + hi * 8;
        float inv = 1.0f / fmaxf(cnt[node], 1.0f);
        float v0 = agg[(size_t)node * DFEAT + m]      * inv + c0[j] + bias[m];
        float v1 = agg[(size_t)node * DFEAT + m + 16] * inv + c1[j] + bias[m + 16];
        v0 = v0 > 0.f ? v0 : (__expf(v0) - 1.f);    // ELU
        v1 = v1 > 0.f ? v1 : (__expf(v1) - 1.f);
        hout[(size_t)node * DFEAT + m]      = v0;
        hout[(size_t)node * DFEAT + m + 16] = v1;
    }
}

// ---------------------------------------------------------------------------
// Kernel 4: fused MLP:  out = ReLU( ReLU(h@mlp1 + b1) @ mlp2 + b2 )
//   D-fragment -> A-fragment re-layout between GEMMs goes through a
//   per-wave LDS slice (1KB of the 320KB/WGP LDS).
// ---------------------------------------------------------------------------
__global__ void __launch_bounds__(64)
mlp_kernel(const float* __restrict__ h, const float* __restrict__ w1,
           const float* __restrict__ b1, const float* __restrict__ w2,
           const float* __restrict__ b2, float* __restrict__ out,
           int nTiles, int C) {
    __shared__ _Float16 lds[2][16 * DFEAT];     // 2 waves / 64-thread block
    int waveInBlk = threadIdx.x >> 5;
    int wave = blockIdx.x * 2 + waveInBlk;
    int lane = threadIdx.x & 31;
    int m = lane & 15, hi = lane >> 4;
    bool active = (wave < nTiles);              // wave-uniform predicate
    int node0 = wave * 16;

    if (active) {
        v16h a   = load_a_f32(h, node0, lane);
        v16h blo = load_b_f32(w1, DFEAT, 0, lane);
        v16h bhi = load_b_f32(w1, DFEAT, 16, lane);
        v8f c0 = {0.f, 0.f, 0.f, 0.f, 0.f, 0.f, 0.f, 0.f};
        v8f c1 = {0.f, 0.f, 0.f, 0.f, 0.f, 0.f, 0.f, 0.f};
        c0 = wmma16(a, blo, c0);
        c1 = wmma16(a, bhi, c1);
#pragma unroll
        for (int j = 0; j < 8; ++j) {           // ReLU+bias, stash tile in LDS (f16)
            int row = j + hi * 8;
            lds[waveInBlk][row * DFEAT + m]      = (_Float16)fmaxf(c0[j] + b1[m], 0.f);
            lds[waveInBlk][row * DFEAT + m + 16] = (_Float16)fmaxf(c1[j] + b1[m + 16], 0.f);
        }
    }
    __syncthreads();
    if (active) {
        v16h a2;                                 // reload in A-fragment layout
#pragma unroll
        for (int j = 0; j < 8; ++j)
#pragma unroll
            for (int hh = 0; hh < 2; ++hh)
                a2[2 * j + hh] = lds[waveInBlk][m * DFEAT + a_kidx(j, hh, hi)];
        v16h bf = load_b_f32(w2, C, 0, lane);    // 32 x 16 -> single B fragment
        v8f o = {0.f, 0.f, 0.f, 0.f, 0.f, 0.f, 0.f, 0.f};
        o = wmma16(a2, bf, o);
#pragma unroll
        for (int j = 0; j < 8; ++j) {
            int node = node0 + j + hi * 8;
            out[(size_t)node * C + m] = fmaxf(o[j] + b2[m], 0.f);
        }
    }
}

// ---------------------------------------------------------------------------
extern "C" void kernel_launch(void* const* d_in, const int* in_sizes, int n_in,
                              void* d_out, int out_size, void* d_ws, size_t ws_size,
                              hipStream_t stream) {
    (void)n_in; (void)out_size; (void)ws_size;

    const float* x     = (const float*)d_in[0];
    const int*   eidx  = (const int*)  d_in[1];
    const float* eattr = (const float*)d_in[2];
    const float* W1    = (const float*)d_in[3];
    const float* root1 = (const float*)d_in[4];
    const float* bias1 = (const float*)d_in[5];
    const float* W2    = (const float*)d_in[6];
    const float* root2 = (const float*)d_in[7];
    const float* bias2 = (const float*)d_in[8];
    const float* mlp1w = (const float*)d_in[9];
    const float* mlp1b = (const float*)d_in[10];
    const float* mlp2w = (const float*)d_in[11];
    const float* mlp2b = (const float*)d_in[12];
    float* out = (float*)d_out;

    const int N  = in_sizes[0] / DFEAT;           // 100000
    const int E  = in_sizes[2] / 2;               // 1600000
    const int K1 = in_sizes[3] / (DFEAT * DFEAT); // 9
    const int K2 = in_sizes[6] / (DFEAT * DFEAT); // 25
    const int C  = in_sizes[12];                  // 16
    const int Kmax = (K2 > K1) ? K2 : K1;
    const int ks1 = (int)(sqrtf((float)K1) + 0.5f);  // 3
    const int ks2 = (int)(sqrtf((float)K2) + 0.5f);  // 5
    const int nTiles = (N + 15) / 16;             // 6250 (exact)

    // Workspace layout (floats): y | agg | cnt | h1 | h2   (~359 MB total)
    float* y   = (float*)d_ws;
    float* agg = y   + (size_t)N * Kmax * DFEAT;
    float* cnt = agg + (size_t)N * DFEAT;
    float* h1  = cnt + (size_t)N;
    float* h2  = h1  + (size_t)N * DFEAT;

    const int* srcA = eidx;
    const int* dstA = eidx + E;

    int gTiles = (nTiles + 1) / 2;   // 2 waves per 64-thread block
    int gEdges = (E + 3) / 4;        // 4 waves per 128-thread block
    size_t aggBytes = ((size_t)N * DFEAT + (size_t)N) * sizeof(float); // agg+cnt contiguous

    // ---- conv1 (ksize=3, K=9) ----
    spline_y_kernel<<<gTiles, 64, 0, stream>>>(x, W1, y, nTiles, K1);
    hipMemsetAsync(agg, 0, aggBytes, stream);
    edge_scatter_kernel<<<gEdges, 128, 0, stream>>>(srcA, dstA, eattr, y, agg, cnt, E, K1, ks1);
    node_update_kernel<<<gTiles, 64, 0, stream>>>(x, root1, bias1, agg, cnt, h1, nTiles);

    // ---- conv2 (ksize=5, K=25) ----
    spline_y_kernel<<<gTiles, 64, 0, stream>>>(h1, W2, y, nTiles, K2);
    hipMemsetAsync(agg, 0, aggBytes, stream);
    edge_scatter_kernel<<<gEdges, 128, 0, stream>>>(srcA, dstA, eattr, y, agg, cnt, E, K2, ks2);
    node_update_kernel<<<gTiles, 64, 0, stream>>>(h1, root2, bias2, agg, cnt, h2, nTiles);

    // ---- fused MLP ----
    mlp_kernel<<<gTiles, 64, 0, stream>>>(h2, mlp1w, mlp1b, mlp2w, mlp2b, out, nTiles, C);
}
